// ScoreModelSE3New_50749333569957
// MI455X (gfx1250) — compile-verified, matrix operand order checked by reading, not run on hardware
//
#include <hip/hip_runtime.h>
#include <hip/hip_fp16.h>
#include <math.h>

typedef __attribute__((ext_vector_type(16))) _Float16 v16h;
typedef __attribute__((ext_vector_type(8)))  _Float16 v8h;
typedef __attribute__((ext_vector_type(8)))  float    v8f;

#define NN  10000
#define GG  200
#define ELN 160000
#define EGN 500000
#define SS  256
#define NAD 16
#define NBD 5

// ---- d_out offsets (float elements), tuple return order ----
#define O_CPRED 0
#define O_CEPS0 30000
#define O_APRED 60000
#define O_AEPS  220000
#define O_BPRED 380000
#define O_BEPS  2880000
#define O_DL    5380000
#define O_RNL   5540000
#define O_AG    6020000
#define O_RNG   6520000

// ---- workspace layout ----
// f32 region (float offsets)
#define W_SUMS1 0
#define W_CNT1  600
#define W_SUMS2 800
#define W_POS1  1400
#define W_WLAST 31400
#define W_S2F   31656
#define WS_F32_FLOATS 2591656
#define WS_H_OFF ((size_t)WS_F32_FLOATS * 4)   // byte offset of f16 region (16B aligned)
// f16 region (half offsets)
#define H_S2H   0
#define H_WB0T  2560000

__device__ __forceinline__ v16h cat16(v8h lo, v8h hi) {
    return __builtin_shufflevector(lo, hi, 0,1,2,3,4,5,6,7,8,9,10,11,12,13,14,15);
}
__device__ __forceinline__ float silu_f(float h) { return h / (1.0f + __expf(-h)); }

// ---------- weight prep: transpose W_b0[257][256] -> wb0t[256][256] f16, last row f32 ----------
__global__ void k_prep_w(const float* __restrict__ Wb0, _Float16* __restrict__ wb0t,
                         float* __restrict__ wlast) {
    int n = blockIdx.x;   // output column 0..255
    int k = threadIdx.x;  // input row 0..255
    wb0t[n * 256 + k] = (_Float16)Wb0[k * 256 + n];
    if (k == 0) wlast[n] = Wb0[256 * 256 + n];
}

__global__ void k_zero(float* __restrict__ p, int n) {
    int i = blockIdx.x * blockDim.x + threadIdx.x;
    if (i < n) p[i] = 0.0f;
}

// ---------- scatter-mean pass 1: sums + counts ----------
__global__ void k_scat1(const float* __restrict__ pos, const int* __restrict__ batch,
                        float* __restrict__ sums1, float* __restrict__ cnt1) {
    int n = blockIdx.x * blockDim.x + threadIdx.x;
    if (n >= NN) return;
    int b = batch[n];
    atomicAdd(&sums1[b * 3 + 0], pos[n * 3 + 0]);
    atomicAdd(&sums1[b * 3 + 1], pos[n * 3 + 1]);
    atomicAdd(&sums1[b * 3 + 2], pos[n * 3 + 2]);
    atomicAdd(&cnt1[b], 1.0f);
}

// ---------- center once -> pos1; accumulate sums for 2nd centering ----------
__global__ void k_cent1(const float* __restrict__ pos, const int* __restrict__ batch,
                        const float* __restrict__ sums1, const float* __restrict__ cnt1,
                        float* __restrict__ pos1, float* __restrict__ sums2) {
    int n = blockIdx.x * blockDim.x + threadIdx.x;
    if (n >= NN) return;
    int b = batch[n];
    float c = fmaxf(cnt1[b], 1.0f);
#pragma unroll
    for (int k = 0; k < 3; ++k) {
        float p = pos[n * 3 + k] - sums1[b * 3 + k] / c;
        pos1[n * 3 + k] = p;
        atomicAdd(&sums2[b * 3 + k], p);
    }
}

// ---------- second centering -> coords_pred; coords_eps0 = 0 ----------
__global__ void k_cent2(const float* __restrict__ pos1, const int* __restrict__ batch,
                        const float* __restrict__ sums2, const float* __restrict__ cnt1,
                        float* __restrict__ out) {
    int n = blockIdx.x * blockDim.x + threadIdx.x;
    if (n >= NN) return;
    int b = batch[n];
    float c = fmaxf(cnt1[b], 1.0f);
#pragma unroll
    for (int k = 0; k < 3; ++k) {
        out[O_CPRED + n * 3 + k] = pos1[n * 3 + k] - sums2[b * 3 + k] / c;
        out[O_CEPS0 + n * 3 + k] = 0.0f;
    }
}

// ---------- fused node MLP: s2 = silu(((x@Wa+ba+tnode)@Wat+bat)@Wsh+bsh); atoms ----------
__global__ __launch_bounds__(256) void k_node(
    const float* __restrict__ x, const float* __restrict__ t, const int* __restrict__ batch,
    const float* __restrict__ W_time, const float* __restrict__ b_time,
    const float* __restrict__ W_atom, const float* __restrict__ b_atom,
    const float* __restrict__ W_at, const float* __restrict__ b_at,
    const float* __restrict__ W_shared, const float* __restrict__ b_shared,
    const float* __restrict__ W_atoms, const float* __restrict__ b_atoms,
    float* __restrict__ s2f, _Float16* __restrict__ s2h, float* __restrict__ out) {
    __shared__ float ldsX[NAD];
    __shared__ float lds0[SS];
    __shared__ float lds1[SS];
    int n = blockIdx.x;
    int c = threadIdx.x;
    if (c < NAD) ldsX[c] = x[n * NAD + c];
    __syncthreads();
    float tval = t[batch[n]];
    float s0 = b_atom[c] + b_time[c] + tval * W_time[c];
#pragma unroll 4
    for (int k = 0; k < NAD; ++k) s0 += ldsX[k] * W_atom[k * SS + c];
    lds0[c] = s0;
    __syncthreads();
    float s1 = b_at[c];
#pragma unroll 4
    for (int k = 0; k < SS; ++k) s1 += lds0[k] * W_at[k * SS + c];
    lds1[c] = s1;
    __syncthreads();
    float h = b_shared[c];
#pragma unroll 4
    for (int k = 0; k < SS; ++k) h += lds1[k] * W_shared[k * SS + c];
    float s2 = silu_f(h);
    s2f[(size_t)n * SS + c] = s2;
    s2h[(size_t)n * SS + c] = (_Float16)s2;
    __syncthreads();
    lds0[c] = s2;   // reuse lds0 for s2
    __syncthreads();
    if (c < 2 * NAD) {
        float av = b_atoms[c];
#pragma unroll 4
        for (int k = 0; k < SS; ++k) av += lds0[k] * W_atoms[k * (2 * NAD) + c];
        if (c < NAD) out[O_AEPS + n * NAD + c] = av;          // atoms[:, :NA] -> eps
        else         out[O_APRED + n * NAD + (c - NAD)] = av; // atoms[:, NA:] -> pred
    }
}

// ---------- local edge attrs: d_l, rn_l ----------
__global__ void k_eattr_l(const int* __restrict__ eil, const float* __restrict__ pos1,
                          float* __restrict__ out) {
    int e = blockIdx.x * blockDim.x + threadIdx.x;
    if (e >= ELN) return;
    int src = eil[e], tgt = eil[ELN + e];
    float rx = pos1[tgt * 3 + 0] - pos1[src * 3 + 0];
    float ry = pos1[tgt * 3 + 1] - pos1[src * 3 + 1];
    float rz = pos1[tgt * 3 + 2] - pos1[src * 3 + 2];
    float d = sqrtf(fmaxf(rx * rx + ry * ry + rz * rz, 1e-6f));
    out[O_DL + e] = d;
    out[O_RNL + e * 3 + 0] = rx / d;
    out[O_RNL + e * 3 + 1] = ry / d;
    out[O_RNL + e * 3 + 2] = rz / d;
}

// ---------- global edge attrs: a_g, rn_g ----------
__global__ void k_eattr_g(const int* __restrict__ eig, const float* __restrict__ pos1,
                          float* __restrict__ out) {
    int e = blockIdx.x * blockDim.x + threadIdx.x;
    if (e >= EGN) return;
    int src = eig[e], tgt = eig[EGN + e];
    float sx = pos1[src * 3 + 0], sy = pos1[src * 3 + 1], sz = pos1[src * 3 + 2];
    float tx = pos1[tgt * 3 + 0], ty = pos1[tgt * 3 + 1], tz = pos1[tgt * 3 + 2];
    float rx = tx - sx, ry = ty - sy, rz = tz - sz;
    float d = sqrtf(fmaxf(rx * rx + ry * ry + rz * rz, 1e-6f));
    out[O_AG + e] = tx * sx + ty * sy + tz * sz;
    out[O_RNG + e * 3 + 0] = rx / d;
    out[O_RNG + e * 3 + 1] = ry / d;
    out[O_RNG + e * 3 + 2] = rz / d;
}

// ---------- edge MLP (dominant): bonds = silu([s_i+s_j, d] @ W_b0 + b_b0) @ W_b1 + b_b1 ----
// Block = 64 threads = 2 waves; each wave handles TWO 16-edge tiles (32 edges) so every
// B fragment fetched from L2 feeds two WMMAs. A fragments for both tiles live in VGPRs;
// the LDS A-staging buffer is then reused as the f16 hidden buffer for the 256->10 GEMM.
__global__ __launch_bounds__(64) void k_edge(
    const int* __restrict__ eig, const float* __restrict__ pos1,
    const _Float16* __restrict__ s2h, const _Float16* __restrict__ wb0t,
    const float* __restrict__ wlast, const float* __restrict__ b_b0,
    const float* __restrict__ W_b1, const float* __restrict__ b_b1,
    float* __restrict__ bonds_pred, float* __restrict__ bonds_eps) {
    __shared__ alignas(16) _Float16 ldsA[2][2][16 * SS]; // [wave][mtile][16x256]; reused as hidden
    __shared__ float ldsD[2][32];
    __shared__ float ldsW1[SS * 2 * NBD];
    __shared__ float ldsB1[2 * NBD];

    for (int idx = threadIdx.x; idx < SS * 2 * NBD; idx += 64) ldsW1[idx] = W_b1[idx];
    if (threadIdx.x < 2 * NBD) ldsB1[threadIdx.x] = b_b1[threadIdx.x];
    __syncthreads();

    const int wave = threadIdx.x >> 5;
    const int lane = threadIdx.x & 31;
    const int e0 = (blockIdx.x * 2 + wave) * 32;
    if (e0 >= EGN) return;
    const int m = lane & 15;     // edge row within a 16-tile
    const int hf = lane >> 4;    // half (selects K sub-range per WMMA A/B layout)

    // ---- stage A tiles (s_i + s_j) and distances ----
#pragma unroll
    for (int mt = 0; mt < 2; ++mt) {
        int e = e0 + mt * 16 + m;
        if (e >= EGN) e = EGN - 1;
        const int j = eig[e];
        const int i = eig[EGN + e];
        const v8h* si = (const v8h*)(s2h + (size_t)i * SS);
        const v8h* sj = (const v8h*)(s2h + (size_t)j * SS);
        v8h* arow = (v8h*)(&ldsA[wave][mt][m * SS]);
#pragma unroll
        for (int q = 0; q < 16; ++q) {
            int c8 = hf * 16 + q;               // lane covers cols hf*128 .. hf*128+127
            arow[c8] = si[c8] + sj[c8];
        }
        if (hf == 0) {
            float dx = pos1[i * 3 + 0] - pos1[j * 3 + 0];
            float dy = pos1[i * 3 + 1] - pos1[j * 3 + 1];
            float dz = pos1[i * 3 + 2] - pos1[j * 3 + 2];
            ldsD[wave][mt * 16 + m] = sqrtf(dx * dx + dy * dy + dz * dz); // unclipped
        }
    }
    asm volatile("s_wait_dscnt 0" ::: "memory");

    // per-lane d values for accumulator rows (C layout: row = r + 8*hf)
    float df[2][8];
#pragma unroll
    for (int mt = 0; mt < 2; ++mt)
#pragma unroll
        for (int r = 0; r < 8; ++r) df[mt][r] = ldsD[wave][mt * 16 + hf * 8 + r];

    // hoist all A fragments into registers (2 tiles x 8 K-tiles)
    v16h afr[2][8];
#pragma unroll
    for (int mt = 0; mt < 2; ++mt)
#pragma unroll
        for (int kt = 0; kt < 8; ++kt) {
            const v8h* pa = (const v8h*)(&ldsA[wave][mt][m * SS + kt * 32]);
            afr[mt][kt] = cat16(pa[hf], pa[2 + hf]); // K {b..b+7},{b+16..b+23}, b=hf*8
        }
    asm volatile("s_wait_dscnt 0" ::: "memory");   // A regs live; LDS buffer now reusable

    _Float16* hid = &ldsA[wave][0][0];             // 32 rows x 256 cols f16 hidden

    // ---- main GEMM: 16 N-tiles, each B fragment feeds both M-tiles ----
    for (int nt = 0; nt < 16; ++nt) {
        const int n = nt * 16 + m;
        v8f acc0 = {};
        v8f acc1 = {};
#pragma unroll
        for (int kt = 0; kt < 8; ++kt) {
            const v8h* pb = (const v8h*)(wb0t + (size_t)n * SS + kt * 32);
            v16h bfr = cat16(pb[hf], pb[2 + hf]);
            acc0 = __builtin_amdgcn_wmma_f32_16x16x32_f16(
                false, afr[0][kt], false, bfr, (short)0, acc0, false, false);
            acc1 = __builtin_amdgcn_wmma_f32_16x16x32_f16(
                false, afr[1][kt], false, bfr, (short)0, acc1, false, false);
        }
        const float bias = b_b0[n];
        const float wl = wlast[n];
#pragma unroll
        for (int r = 0; r < 8; ++r) {
            float h0 = acc0[r] + bias + df[0][r] * wl;  // exact f32 rank-1 distance term
            float h1 = acc1[r] + bias + df[1][r] * wl;
            hid[(hf * 8 + r) * SS + n]        = (_Float16)silu_f(h0);
            hid[(16 + hf * 8 + r) * SS + n]   = (_Float16)silu_f(h1);
        }
    }
    asm volatile("s_wait_dscnt 0" ::: "memory");

    // ---- second GEMM (256 -> 10), f32 VALU: lane does edge m of each tile, o = hf*5+q ----
#pragma unroll
    for (int mt = 0; mt < 2; ++mt) {
        const int e = e0 + mt * 16 + m;
        float acc2[NBD] = {0.f, 0.f, 0.f, 0.f, 0.f};
        const _Float16* hrow = hid + (mt * 16 + m) * SS;
        for (int k = 0; k < SS; ++k) {
            float hv = (float)hrow[k];
            const float* wr = &ldsW1[k * (2 * NBD) + hf * NBD];
#pragma unroll
            for (int q = 0; q < NBD; ++q) acc2[q] += hv * wr[q];
        }
        if (e < EGN) {
#pragma unroll
            for (int q = 0; q < NBD; ++q) {
                int o = hf * NBD + q;
                float val = acc2[q] + ldsB1[o];
                if (o < NBD) bonds_pred[(size_t)e * NBD + o] = val;
                else         bonds_eps[(size_t)e * NBD + (o - NBD)] = val;
            }
        }
    }
}

extern "C" void kernel_launch(void* const* d_in, const int* in_sizes, int n_in,
                              void* d_out, int out_size, void* d_ws, size_t ws_size,
                              hipStream_t stream) {
    // inputs in setup_inputs() dict order
    const float* x        = (const float*)d_in[0];
    const float* t        = (const float*)d_in[1];
    const float* pos      = (const float*)d_in[2];
    const int*   eil      = (const int*)d_in[3];
    const int*   eig      = (const int*)d_in[4];
    // d_in[5] = edge_attr_global (unused by reference output)
    const int*   batch    = (const int*)d_in[6];
    const float* W_time   = (const float*)d_in[7];
    const float* b_time   = (const float*)d_in[8];
    const float* W_atom   = (const float*)d_in[9];
    const float* b_atom   = (const float*)d_in[10];
    const float* W_at     = (const float*)d_in[11];
    const float* b_at     = (const float*)d_in[12];
    const float* W_shared = (const float*)d_in[13];
    const float* b_shared = (const float*)d_in[14];
    const float* W_b0     = (const float*)d_in[15];
    const float* b_b0     = (const float*)d_in[16];
    const float* W_b1     = (const float*)d_in[17];
    const float* b_b1     = (const float*)d_in[18];
    // d_in[19] = W_coord (multiplies v == 0; unused)
    const float* W_atoms  = (const float*)d_in[20];
    const float* b_atoms  = (const float*)d_in[21];

    float* out = (float*)d_out;
    float* F   = (float*)d_ws;
    _Float16* H = (_Float16*)((char*)d_ws + WS_H_OFF);

    float*    sums1 = F + W_SUMS1;
    float*    cnt1  = F + W_CNT1;
    float*    sums2 = F + W_SUMS2;
    float*    pos1  = F + W_POS1;
    float*    wlast = F + W_WLAST;
    float*    s2f   = F + W_S2F;
    _Float16* s2h   = H + H_S2H;
    _Float16* wb0t  = H + H_WB0T;

    // weight prep + zero accumulators
    k_prep_w<<<SS, SS, 0, stream>>>(W_b0, wb0t, wlast);
    k_zero<<<6, 256, 0, stream>>>(F + W_SUMS1, 1400);

    // pos centering chain
    int nb = (NN + 255) / 256;
    k_scat1<<<nb, 256, 0, stream>>>(pos, batch, sums1, cnt1);
    k_cent1<<<nb, 256, 0, stream>>>(pos, batch, sums1, cnt1, pos1, sums2);
    k_cent2<<<nb, 256, 0, stream>>>(pos1, batch, sums2, cnt1, out);

    // node MLP (writes s2 f32/f16 + atoms outputs)
    k_node<<<NN, SS, 0, stream>>>(x, t, batch, W_time, b_time, W_atom, b_atom,
                                  W_at, b_at, W_shared, b_shared, W_atoms, b_atoms,
                                  s2f, s2h, out);

    // edge attrs
    k_eattr_l<<<(ELN + 255) / 256, 256, 0, stream>>>(eil, pos1, out);
    k_eattr_g<<<(EGN + 255) / 256, 256, 0, stream>>>(eig, pos1, out);

    // dominant WMMA edge MLP: 64 edges/block (2 waves x 2 tiles)
    k_edge<<<(EGN + 63) / 64, 64, 0, stream>>>(eig, pos1, s2h, wb0t, wlast,
                                               b_b0, W_b1, b_b1,
                                               out + O_BPRED, out + O_BEPS);
}
